// MAGNN_ctr_ntype_specific_6889127542847
// MI455X (gfx1250) — matrix operand order repaired and checked by reading, not compiled, not run on hardware
//
#include <hip/hip_runtime.h>

// ---------------------------------------------------------------------------
// Problem constants (match the reference)
// ---------------------------------------------------------------------------
#define N_NODES   20000
#define D_FEAT    256
#define P_PATHS   3
#define E_EDGES   60000
#define L_SEQ     4
#define H_HEADS   8
#define D_HEAD    32
#define D_MODEL   256      // H*D
#define G3        768      // 3*D_MODEL

// ---------------------------------------------------------------------------
// CDNA5 WMMA types
// ---------------------------------------------------------------------------
typedef __attribute__((ext_vector_type(16))) __bf16 v16bf;
typedef __attribute__((ext_vector_type(8)))  float  v8f;

// ---------------------------------------------------------------------------
// bf16 helpers (stored as unsigned short bit patterns)
// ---------------------------------------------------------------------------
__device__ __forceinline__ unsigned short f32_to_bf16(float f) {
  unsigned int u = __float_as_uint(f);
  u += 0x7FFFu + ((u >> 16) & 1u);           // round-to-nearest-even
  return (unsigned short)(u >> 16);
}
__device__ __forceinline__ float bf16_to_f32(unsigned short h) {
  return __uint_as_float(((unsigned int)h) << 16);
}

// ---------------------------------------------------------------------------
// CDNA5 async global->LDS copy (ASYNCcnt tracked, no VGPR data round-trip).
// VDST VGPR holds the LDS byte address (= low 32 bits of generic pointer per
// the ISA flat->LDS mapping); VADDR holds the 64-bit global address.
// ---------------------------------------------------------------------------
__device__ __forceinline__ void async_load_b128(void* lds_ptr, const void* gptr) {
  unsigned lds_addr = (unsigned)(unsigned long long)lds_ptr;
  asm volatile("global_load_async_to_lds_b128 %0, %1, off"
               :: "v"(lds_addr), "v"(gptr)
               : "memory");
}
__device__ __forceinline__ void wait_async0() {
  asm volatile("s_wait_asynccnt 0" ::: "memory");
}

// ---------------------------------------------------------------------------
// Fragment loaders (wave32): contiguous k order, same for A and B (not the ISA table order).
// A (16x32, MxK):  lane m = lane&15, hi = lane>>4; element e -> K = k0+hi*16+e
//   => one contiguous 32B run at (k0 + hi*16)
// B (32x16, KxN): lane n = lane&15, hi = lane>>4; element e -> K = k0+hi*16+e
//   => one contiguous 32B run at row j, (k0 + hi*16)
// ---------------------------------------------------------------------------
__device__ __forceinline__ v16bf load_a_frag(const unsigned short* arow, int k0, int hi) {
  union { uint4 u[2]; v16bf v; } t;
  t.u[0] = *(const uint4*)(arow + k0 + hi * 16);
  t.u[1] = *(const uint4*)(arow + k0 + hi * 16 + 8);
  return t.v;
}
__device__ __forceinline__ v16bf load_b_frag(const unsigned short* wrow, int k0, int hi) {
  union { uint4 u[2]; v16bf v; } t;
  const uint4* pp = (const uint4*)(wrow + k0 + hi * 16);
  t.u[0] = pp[0];
  t.u[1] = pp[1];
  return t.v;
}
__device__ __forceinline__ v8f wmma_bf16(v16bf a, v16bf b, v8f c) {
  return __builtin_amdgcn_wmma_f32_16x16x32_bf16(false, a, false, b, (short)0, c,
                                                 false, false);
}

// ---------------------------------------------------------------------------
// Utility kernels
// ---------------------------------------------------------------------------
__global__ void zero_f32_kernel(float* __restrict__ p, long long count) {
  long long i = (long long)blockIdx.x * blockDim.x + threadIdx.x;
  if (i < count) p[i] = 0.0f;
}

__global__ void cvt_bf16_kernel(const float* __restrict__ src,
                                unsigned short* __restrict__ dst, long long count) {
  long long i = (long long)blockIdx.x * blockDim.x + threadIdx.x;
  if (i < count) dst[i] = f32_to_bf16(src[i]);
}

// l2-normalize features across heads: ft[n,h,d] = x[n,h,d]/max(||x[n,:,d]||,eps)
// one wave per node; lane = d index; the 8-head vector lives entirely in-lane.
__global__ __launch_bounds__(256) void ft_norm_kernel(const float* __restrict__ feat,
                                                      float* __restrict__ ftn) {
  int wave = (blockIdx.x * blockDim.x + threadIdx.x) >> 5;
  int lane = threadIdx.x & 31;
  if (wave >= N_NODES) return;
  const float* row = feat + (size_t)wave * D_MODEL;
  float x[H_HEADS], ss = 0.0f;
#pragma unroll
  for (int h = 0; h < H_HEADS; ++h) { x[h] = row[h * D_HEAD + lane]; ss += x[h] * x[h]; }
  float inv = 1.0f / fmaxf(__builtin_sqrtf(ss), 1e-12f);
  float* orow = ftn + (size_t)wave * D_MODEL;
#pragma unroll
  for (int h = 0; h < H_HEADS; ++h) orow[h * D_HEAD + lane] = x[h] * inv;
}

// ---------------------------------------------------------------------------
// Fused GRU step:
//   gi = gather(x_t) @ Wih^T + b_ih ; gh = h_prev @ Whh^T + b_hh
//   h  = (1-z)*tanh(i_n + r*h_n) + z*h_prev
// Block: 256 threads (8 waves). Block tile: 32 rows x 128 cols; x_t (gathered)
// and h_prev rows staged in LDS via async global->LDS b128 copies. Each wave
// owns one 16-col tile and TWO 16-row tiles, reusing every B (weight) fragment
// for both row tiles: 12 WMMAs per 6 B-fragment loads per K-step.
// Grid: (2 col groups, E/32 row tiles).
// ---------------------------------------------------------------------------
__global__ __launch_bounds__(256) void gru_step_kernel(
    const unsigned short* __restrict__ featB,   // [N, 256] bf16
    const unsigned short* __restrict__ Wih,     // [768, 256] bf16 (this p)
    const unsigned short* __restrict__ Whh,     // [768, 256] bf16 (this p)
    const float* __restrict__ bih,              // [768] (this p)
    const float* __restrict__ bhh,              // [768] (this p)
    const int* __restrict__ idx,                // [E, L] (this p)
    const unsigned short* __restrict__ hprev,   // [E, 256] bf16
    unsigned short* __restrict__ hnext,         // [E, 256] bf16
    float* __restrict__ hiddenF32,              // [E, 256] f32 (written if wf32)
    int t, int first, int wf32) {
  __shared__ uint4 ldsX4[32 * 32];   // 32 rows x 256 bf16 of gathered x_t
  __shared__ uint4 ldsH4[32 * 32];   // 32 rows x 256 bf16 of h_prev
  unsigned short* ldsX = (unsigned short*)ldsX4;
  unsigned short* ldsH = (unsigned short*)ldsH4;

  const int tid   = threadIdx.x;
  const int lane  = tid & 31;
  const int waveI = tid >> 5;
  const int rowT  = blockIdx.y;                  // 32-row tile index
  const int colBase = blockIdx.x * 128 + waveI * 16;

  // --- async staging: gathered x rows (+ h_prev rows if !first) -------------
  for (int v = tid; v < 32 * 32; v += 256) {     // 1024 x 16B chunks each
    int rr = v >> 5, cc = v & 31;
    int grow = rowT * 32 + rr;
    int src  = idx[(size_t)grow * L_SEQ + t];
    async_load_b128(&ldsX4[rr * 32 + cc],
                    featB + (size_t)src * D_MODEL + cc * 8);
    if (!first)
      async_load_b128(&ldsH4[rr * 32 + cc],
                      hprev + (size_t)grow * D_MODEL + cc * 8);
  }
  wait_async0();
  __syncthreads();

  const int m16 = lane & 15;       // A-matrix row / B-matrix column within tile
  const int hi  = lane >> 4;

  v8f accI[3][2] = {};   // [gate r/z/n][row tile 0/1], input side
  v8f accH[3][2] = {};   // hidden side

  const unsigned short* arowX0 = ldsX + m16 * D_MODEL;
  const unsigned short* arowX1 = ldsX + (m16 + 16) * D_MODEL;
  const unsigned short* arowH0 = ldsH + m16 * D_MODEL;
  const unsigned short* arowH1 = ldsH + (m16 + 16) * D_MODEL;
  const int col = colBase + m16;   // this lane's output column / W row offset

#pragma unroll
  for (int k0 = 0; k0 < D_MODEL; k0 += 32) {
    v16bf ax0 = load_a_frag(arowX0, k0, hi);
    v16bf ax1 = load_a_frag(arowX1, k0, hi);
#pragma unroll
    for (int g = 0; g < 3; ++g) {
      v16bf bI = load_b_frag(Wih + (size_t)(g * D_MODEL + col) * D_MODEL, k0, hi);
      accI[g][0] = wmma_bf16(ax0, bI, accI[g][0]);
      accI[g][1] = wmma_bf16(ax1, bI, accI[g][1]);
    }
    if (!first) {
      v16bf ah0 = load_a_frag(arowH0, k0, hi);
      v16bf ah1 = load_a_frag(arowH1, k0, hi);
#pragma unroll
      for (int g = 0; g < 3; ++g) {
        v16bf bH = load_b_frag(Whh + (size_t)(g * D_MODEL + col) * D_MODEL, k0, hi);
        accH[g][0] = wmma_bf16(ah0, bH, accH[g][0]);
        accH[g][1] = wmma_bf16(ah1, bH, accH[g][1]);
      }
    }
  }

  // --- epilogue: gate math; D layout lane n=lane&15, m = i + 8*hi -----------
  const float bir = bih[col],           biz = bih[D_MODEL + col],
              bin = bih[2 * D_MODEL + col];
  const float bhr = bhh[col],           bhz = bhh[D_MODEL + col],
              bhn = bhh[2 * D_MODEL + col];
#pragma unroll
  for (int rt = 0; rt < 2; ++rt) {
#pragma unroll
    for (int i = 0; i < 8; ++i) {
      int m    = i + 8 * hi + rt * 16;
      int grow = rowT * 32 + m;
      float ir = accI[0][rt][i] + bir, iz = accI[1][rt][i] + biz,
            inn = accI[2][rt][i] + bin;
      float hr = accH[0][rt][i] + bhr, hz = accH[1][rt][i] + bhz,
            hn  = accH[2][rt][i] + bhn;
      float rg = 1.0f / (1.0f + __expf(-(ir + hr)));
      float zg = 1.0f / (1.0f + __expf(-(iz + hz)));
      float ng = tanhf(inn + rg * hn);
      float hp = first ? 0.0f : bf16_to_f32(ldsH[m * D_MODEL + col]);
      float hv = (1.0f - zg) * ng + zg * hp;
      hnext[(size_t)grow * D_MODEL + col] = f32_to_bf16(hv);
      if (wf32) hiddenF32[(size_t)grow * D_MODEL + col] = hv;
    }
  }
}

// ---------------------------------------------------------------------------
// Per-edge head attention + segment-sum: one wave per edge, lane = d index.
// eft = l2norm over heads (in-lane); sim[h] reduced over 32 lanes (d);
// softmax over 8 heads; atomic segment-sum into nft[p].
// ---------------------------------------------------------------------------
__global__ __launch_bounds__(256) void edge_attn_kernel(
    const float* __restrict__ hiddenF32,   // [E, 256]
    const float* __restrict__ ftn,         // [N, 256] normalized features
    const int*   __restrict__ dst,         // [E] (this p)
    float*       __restrict__ nft) {       // [N, 256] (this p), zero-initialized
  int e    = (blockIdx.x * blockDim.x + threadIdx.x) >> 5;
  int lane = threadIdx.x & 31;
  if (e >= E_EDGES) return;

  const float* hrow = hiddenF32 + (size_t)e * D_MODEL;
  float x[H_HEADS], ss = 0.0f;
#pragma unroll
  for (int h = 0; h < H_HEADS; ++h) { x[h] = hrow[h * D_HEAD + lane]; ss += x[h] * x[h]; }
  float inv = 1.0f / fmaxf(__builtin_sqrtf(ss), 1e-12f);

  int dn = dst[e];
  const float* frow = ftn + (size_t)dn * D_MODEL;
  float sim[H_HEADS];
#pragma unroll
  for (int h = 0; h < H_HEADS; ++h) {
    x[h] *= inv;
    sim[h] = x[h] * frow[h * D_HEAD + lane];
  }
#pragma unroll
  for (int h = 0; h < H_HEADS; ++h)
#pragma unroll
    for (int off = 16; off > 0; off >>= 1) sim[h] += __shfl_xor(sim[h], off, 32);

  float mx = sim[0];
#pragma unroll
  for (int h = 1; h < H_HEADS; ++h) mx = fmaxf(mx, sim[h]);
  float den = 0.0f, a[H_HEADS];
#pragma unroll
  for (int h = 0; h < H_HEADS; ++h) { a[h] = __expf(sim[h] - mx); den += a[h]; }
  float rden = 1.0f / den;

  float* nrow = nft + (size_t)dn * D_MODEL;
#pragma unroll
  for (int h = 0; h < H_HEADS; ++h)
    atomicAdd(nrow + h * D_HEAD + lane, x[h] * a[h] * rden);
}

// ---------------------------------------------------------------------------
// Semantic attention scores: one wave per (p,n); lane = e index (0..31).
// s[p] += mean_n sum_{h,e} tanh( h[p,n,h,:] . fc1_w[e,:] ) * fc2[h,e]
// ---------------------------------------------------------------------------
__global__ __launch_bounds__(256) void sem_score_kernel(
    const float* __restrict__ nft,   // [P, N, 256]
    const float* __restrict__ fc1,   // [32, 32]
    const float* __restrict__ fc2,   // [8, 32]
    float*       __restrict__ sbuf) {
  int gid  = (blockIdx.x * blockDim.x + threadIdx.x) >> 5;
  int lane = threadIdx.x & 31;
  if (gid >= P_PATHS * N_NODES) return;
  int p = gid / N_NODES, n = gid - p * N_NODES;

  const float* hrow = nft + ((size_t)p * N_NODES + n) * D_MODEL;
  const float* w    = fc1 + lane * D_HEAD;   // fc1_w[e=lane, :]
  float acc = 0.0f;
#pragma unroll
  for (int h = 0; h < H_HEADS; ++h) {
    float v = 0.0f;
#pragma unroll
    for (int d = 0; d < D_HEAD; ++d) v += hrow[h * D_HEAD + d] * w[d];
    acc += tanhf(v) * fc2[h * D_HEAD + lane];
  }
#pragma unroll
  for (int off = 16; off > 0; off >>= 1) acc += __shfl_xor(acc, off, 32);
  if (lane == 0) atomicAdd(&sbuf[p], acc * (1.0f / (float)N_NODES));
}

// Final: beta = softmax(s) (recomputed per thread), out = sum_p beta_p*nft_p
__global__ void final_out_kernel(const float* __restrict__ nft,
                                 const float* __restrict__ sbuf,
                                 float* __restrict__ out) {
  long long i = (long long)blockIdx.x * blockDim.x + threadIdx.x;
  const long long total = (long long)N_NODES * D_MODEL;
  if (i >= total) return;
  float s0 = sbuf[0], s1 = sbuf[1], s2 = sbuf[2];
  float mx = fmaxf(s0, fmaxf(s1, s2));
  float e0 = __expf(s0 - mx), e1 = __expf(s1 - mx), e2 = __expf(s2 - mx);
  float r  = 1.0f / (e0 + e1 + e2);
  out[i] = (e0 * nft[i] + e1 * nft[total + i] + e2 * nft[2 * total + i]) * r;
}

// ---------------------------------------------------------------------------
// Host launcher
// ---------------------------------------------------------------------------
extern "C" void kernel_launch(void* const* d_in, const int* in_sizes, int n_in,
                              void* d_out, int out_size, void* d_ws, size_t ws_size,
                              hipStream_t stream) {
  const float* features = (const float*)d_in[0];
  const float* w_ih     = (const float*)d_in[1];   // [P, 768, 256]
  const float* w_hh     = (const float*)d_in[2];   // [P, 768, 256]
  const float* b_ih     = (const float*)d_in[3];   // [P, 768]
  const float* b_hh     = (const float*)d_in[4];   // [P, 768]
  const float* fc1      = (const float*)d_in[5];   // [32, 32]
  const float* fc2      = (const float*)d_in[6];   // [8, 32]
  const int*   empi     = (const int*)d_in[7];     // [P, E, L]
  const int*   edst     = (const int*)d_in[8];     // [P, E]
  float*       out      = (float*)d_out;

  // ---- workspace layout (256B aligned slabs) ------------------------------
  char* ws = (char*)d_ws;
  size_t off = 0;
  auto take = [&](size_t bytes) -> char* {
    char* p = ws + off;
    off += (bytes + 255) & ~(size_t)255;
    return p;
  };
  const size_t PN256 = (size_t)P_PATHS * N_NODES * D_MODEL;
  float*          nft    = (float*)take(PN256 * 4);
  float*          sbuf   = (float*)take(16);
  unsigned short* featB  = (unsigned short*)take((size_t)N_NODES * D_MODEL * 2);
  unsigned short* wihB   = (unsigned short*)take((size_t)P_PATHS * G3 * D_MODEL * 2);
  unsigned short* whhB   = (unsigned short*)take((size_t)P_PATHS * G3 * D_MODEL * 2);
  float*          ftn    = (float*)take((size_t)N_NODES * D_MODEL * 4);
  unsigned short* hA     = (unsigned short*)take((size_t)E_EDGES * D_MODEL * 2);
  unsigned short* hB     = (unsigned short*)take((size_t)E_EDGES * D_MODEL * 2);
  float*          hidden = (float*)take((size_t)E_EDGES * D_MODEL * 4);
  (void)ws_size; (void)n_in; (void)in_sizes; (void)out_size;

  // ---- prep ---------------------------------------------------------------
  {
    long long c = (long long)PN256;
    zero_f32_kernel<<<(unsigned)((c + 255) / 256), 256, 0, stream>>>(nft, c);
    zero_f32_kernel<<<1, 256, 0, stream>>>(sbuf, 4);
    long long cf = (long long)N_NODES * D_MODEL;
    cvt_bf16_kernel<<<(unsigned)((cf + 255) / 256), 256, 0, stream>>>(features, featB, cf);
    long long cw = (long long)P_PATHS * G3 * D_MODEL;
    cvt_bf16_kernel<<<(unsigned)((cw + 255) / 256), 256, 0, stream>>>(w_ih, wihB, cw);
    cvt_bf16_kernel<<<(unsigned)((cw + 255) / 256), 256, 0, stream>>>(w_hh, whhB, cw);
    ft_norm_kernel<<<N_NODES / 8, 256, 0, stream>>>(features, ftn);
  }

  // ---- per-metapath GRU (WMMA) + edge attention ---------------------------
  dim3 gruGrid(2, E_EDGES / 32);   // 2 col groups x 1875 row tiles
  for (int p = 0; p < P_PATHS; ++p) {
    const unsigned short* Wih = wihB + (size_t)p * G3 * D_MODEL;
    const unsigned short* Whh = whhB + (size_t)p * G3 * D_MODEL;
    const float* bihp = b_ih + (size_t)p * G3;
    const float* bhhp = b_hh + (size_t)p * G3;
    const int*   idxp = empi + (size_t)p * E_EDGES * L_SEQ;

    // ping-pong: t0 -> hA, t1 -> hB, t2 -> hA, t3 -> hB (+ f32 hidden)
    unsigned short* bufs[2] = {hA, hB};
    for (int t = 0; t < L_SEQ; ++t) {
      unsigned short* hn = bufs[t & 1];
      unsigned short* hp = bufs[(t + 1) & 1];
      gru_step_kernel<<<gruGrid, 256, 0, stream>>>(
          featB, Wih, Whh, bihp, bhhp, idxp, hp, hn, hidden,
          t, (t == 0) ? 1 : 0, (t == L_SEQ - 1) ? 1 : 0);
    }
    edge_attn_kernel<<<E_EDGES / 8, 256, 0, stream>>>(
        hidden, ftn, edst + (size_t)p * E_EDGES,
        nft + (size_t)p * N_NODES * D_MODEL);
  }

  // ---- semantic attention + output ---------------------------------------
  sem_score_kernel<<<(P_PATHS * N_NODES) / 8, 256, 0, stream>>>(nft, fc1, fc2, sbuf);
  long long ct = (long long)N_NODES * D_MODEL;
  final_out_kernel<<<(unsigned)((ct + 255) / 256), 256, 0, stream>>>(nft, sbuf, out);
}
